// OptimizedConv1DSelfONN_80676665688289
// MI455X (gfx1250) — matrix operand order repaired and checked by reading, not compiled, not run on hardware
//
#include <hip/hip_runtime.h>
#include <hip/hip_bf16.h>
#include <stdint.h>

// ---------------------------------------------------------------------------
// Self-ONN conv1d (B=16, Cin=128, T=8192, Cout=256, K=9, Q=5, pad=4) as
// 9 shifted GEMMs on bf16 WMMA / f32 accum. Weights double-buffered through
// LDS via gfx1250 async-to-LDS loads; fused nonlinear epilogue.
// ---------------------------------------------------------------------------

typedef __attribute__((ext_vector_type(16))) __bf16 v16bf;
typedef __attribute__((ext_vector_type(8)))  float  v8f;

struct Frag32B { uint4 a, b; };   // 32 bytes == one v16bf fragment per lane

__device__ __forceinline__ unsigned short f2bf(float f) {
  unsigned u = __float_as_uint(f);
  u += 0x7FFFu + ((u >> 16) & 1u);      // round-to-nearest-even
  return (unsigned short)(u >> 16);
}

// LDS layout (bytes):
//   [0, 34816)          x slab: [tl 0..135][ci 0..127] bf16, XOR-8 swizzle
//   [34816, 45056)      weight stage buffer 0 (10 fragments x 1024 B)
//   [45056, 55296)      weight stage buffer 1
#define XSLAB_BYTES  34816
#define WBUF_BYTES   10240
#define LDS_TOTAL    55296

// ---------------------------------------------------------------------------
// Kernel 1: repack conv_weights (Cout,Cin,9,5) f32 -> bf16 A-fragments in ws.
// Fragment f = (((ct*9 + k)*4 + cc)*5 + q), 1024 B each, half-split bytes:
//   ushort index within fragment = h*256 + L*8 + jj   (h = j>>3, jj = j&7)
// Element semantics (CDNA5 16-bit A 16x32 layout):
//   M = L & 15            -> co = ct*16 + M
//   K = h*16 + (L>>4)*8 + jj -> ci = cc*32 + K
// ---------------------------------------------------------------------------
__global__ void prep_weights_kernel(const float* __restrict__ w,
                                    unsigned short* __restrict__ wsA) {
  int idx = blockIdx.x * blockDim.x + threadIdx.x;
  if (idx >= 16 * 9 * 4 * 5 * 512) return;
  int jj = idx & 7;
  int L  = (idx >> 3) & 31;
  int h  = (idx >> 8) & 1;
  int f  = idx >> 9;
  int q  = f % 5;
  int f2 = f / 5;
  int cc = f2 & 3;
  int f3 = f2 >> 2;
  int k  = f3 % 9;
  int ct = f3 / 9;
  int co = ct * 16 + (L & 15);
  int ci = cc * 32 + h * 16 + ((L >> 4) * 8) + jj;
  float v = w[(((size_t)co * 128 + ci) * 9 + k) * 5 + q];
  wsA[idx] = f2bf(v);
}

// ---------------------------------------------------------------------------
// Kernel 2: softmax(operator_probs) -> ws (stride 8 floats per channel)
// ---------------------------------------------------------------------------
__global__ void prep_softmax_kernel(const float* __restrict__ p,
                                    float* __restrict__ opw) {
  int c = blockIdx.x * blockDim.x + threadIdx.x;
  if (c >= 256) return;
  float v[5], m = -1e30f;
  for (int q = 0; q < 5; ++q) { v[q] = p[c * 5 + q]; m = fmaxf(m, v[q]); }
  float s = 0.f;
  for (int q = 0; q < 5; ++q) { v[q] = __expf(v[q] - m); s += v[q]; }
  float inv = 1.f / s;
  for (int q = 0; q < 5; ++q) opw[c * 8 + q] = v[q] * inv;
  opw[c * 8 + 5] = 0.f; opw[c * 8 + 6] = 0.f; opw[c * 8 + 7] = 0.f;
}

// ---------------------------------------------------------------------------
// async 16-byte global -> LDS copy (ASYNCcnt-tracked, gfx1250)
// ---------------------------------------------------------------------------
__device__ __forceinline__ void async_cp16(unsigned ldsOff, const char* g) {
  unsigned long long ga = (unsigned long long)(uintptr_t)g;
  asm volatile("global_load_async_to_lds_b128 %0, %1, off"
               :: "v"(ldsOff), "v"(ga) : "memory");
}
__device__ __forceinline__ void wait_async0() {
  asm volatile("s_wait_asynccnt 0x0" ::: "memory");
}

// ---------------------------------------------------------------------------
// Kernel 3: main fused conv + operator mix.
// grid = (64 t-blocks, 16 co-tiles, 16 batches), block = 256 (8 waves).
// 18 stages: stage s covers (k = s>>1, cc pair = s&1), double-buffered.
// ---------------------------------------------------------------------------
__global__ void __launch_bounds__(256)
selfonn_main_kernel(const float* __restrict__ x,
                    const unsigned short* __restrict__ wsA,
                    const float* __restrict__ opw,
                    float* __restrict__ out) {
  __shared__ __align__(16) unsigned char lds[LDS_TOTAL];

  const int tid  = threadIdx.x;
  const int lane = tid & 31;
  const int wave = tid >> 5;           // 0..7 -> 16-col t strip
  const int t0   = blockIdx.x * 128;
  const int ct   = blockIdx.y;         // co tile (16 channels)
  const int b    = blockIdx.z;
  const int hi   = lane >> 4;
  const int n    = lane & 15;

  const unsigned ldsBase = (unsigned)(uintptr_t)&lds[0];  // LDS byte offset
  const char* wsW = (const char*)wsA;

  // --- issue async stage for s=0 (overlaps with x-slab fill below) ---------
  {
    size_t chunk = ((size_t)ct * 9 + 0) * 20480;   // k=0, cc pair 0 -> first 10 KB
    for (int u = tid; u < 640; u += 256)
      async_cp16(ldsBase + XSLAB_BYTES + u * 16, wsW + chunk + (size_t)u * 16);
  }

  // --- stage x[b, 0..127, t0-4 .. t0+131] into LDS as bf16 (swizzled) ------
  unsigned short* ldsX = (unsigned short*)&lds[0];
  const float* xb = x + ((size_t)b * 128) * 8192;
  for (int i = tid; i < 136 * 128; i += 256) {
    int ci = i / 136;
    int tl = i - ci * 136;
    int t  = t0 - 4 + tl;
    float v = 0.f;
    if ((unsigned)t < 8192u) v = xb[(size_t)ci * 8192 + t];
    int pg = (ci >> 3) ^ (tl & 7);
    ldsX[tl * 128 + pg * 8 + (ci & 7)] = f2bf(v);
  }
  wait_async0();
  __syncthreads();

  // --- fragment loaders ----------------------------------------------------
  const char* ldsC = (const char*)&lds[0];
  auto loadB = [&](int k, int cc) -> v16bf {
    int tl = wave * 16 + n + k;
    int sw = tl & 7;
    const char* row = ldsC + tl * 256;
    int g1 = cc * 4 + hi;
    Frag32B f;
    f.a = *(const uint4*)(row + (((g1)     ^ sw) * 16));
    f.b = *(const uint4*)(row + (((g1 + 2) ^ sw) * 16));
    return __builtin_bit_cast(v16bf, f);
  };
  auto loadA = [&](int buf, int fc) -> v16bf {   // fc in 0..9
    const char* p = ldsC + XSLAB_BYTES + buf * WBUF_BYTES + fc * 1024 + lane * 16;
    Frag32B f;
    f.a = *(const uint4*)p;
    f.b = *(const uint4*)(p + 512);
    return __builtin_bit_cast(v16bf, f);
  };

  // --- main loop: 18 stages, weights double-buffered via async-to-LDS ------
  v8f acc[5] = {};
  for (int s = 0; s < 18; ++s) {
    const int k      = s >> 1;
    const int ccbase = (s & 1) * 2;
    const int buf    = s & 1;

    if (s < 17) {   // async-prefetch next stage into the other buffer
      int s1 = s + 1;
      size_t chunk = (((size_t)ct * 9 + (s1 >> 1)) * 2 + (s1 & 1)) * 10240;
      unsigned dst = ldsBase + XSLAB_BYTES + (s1 & 1) * WBUF_BYTES;
      for (int u = tid; u < 640; u += 256)
        async_cp16(dst + u * 16, wsW + chunk + (size_t)u * 16);
    }

    // issue all 24 ds loads for this stage, then 10 WMMAs
    v16bf B0 = loadB(k, ccbase);
    v16bf B1 = loadB(k, ccbase + 1);
    v16bf A0[5], A1[5];
#pragma unroll
    for (int q = 0; q < 5; ++q) A0[q] = loadA(buf, q);
#pragma unroll
    for (int q = 0; q < 5; ++q) A1[q] = loadA(buf, 5 + q);
#pragma unroll
    for (int q = 0; q < 5; ++q)
      acc[q] = __builtin_amdgcn_wmma_f32_16x16x32_bf16(
          false, A0[q], false, B0, (short)0, acc[q], false, false);
#pragma unroll
    for (int q = 0; q < 5; ++q)
      acc[q] = __builtin_amdgcn_wmma_f32_16x16x32_bf16(
          false, A1[q], false, B1, (short)0, acc[q], false, false);

    wait_async0();      // next stage's buffer fully written
    __syncthreads();    // all waves done reading / writing
  }

  // --- epilogue: operator nonlinearities + softmax mix, f32 store ----------
  const int tcol = t0 + wave * 16 + n;
#pragma unroll
  for (int r = 0; r < 8; ++r) {
    int co = ct * 16 + r + hi * 8;
    const float* w = opw + co * 8;
    float w0 = w[0], w1 = w[1], w2 = w[2], w3 = w[3], w4 = w[4];
    float z0 = acc[0][r];
    float z1 = acc[1][r];
    float z2 = acc[2][r];
    float z3 = acc[3][r];
    float z4 = acc[4][r];
    float zc = fminf(fmaxf(z3, -15.f), 15.f);
    float e2 = __expf(2.f * zc);
    float th = (e2 - 1.f) / (e2 + 1.f);
    float ec = __expf(fminf(fmaxf(z4, -8.f), 8.f));
    float res = w0 * z0 + w1 * __sinf(z1) + w2 * __cosf(z2) + w3 * th + w4 * ec;
    out[((size_t)b * 256 + co) * 8192 + tcol] = res;
  }
}

// ---------------------------------------------------------------------------
// Launch
// ---------------------------------------------------------------------------
extern "C" void kernel_launch(void* const* d_in, const int* in_sizes, int n_in,
                              void* d_out, int out_size, void* d_ws, size_t ws_size,
                              hipStream_t stream) {
  (void)in_sizes; (void)n_in; (void)out_size; (void)ws_size;
  const float* x  = (const float*)d_in[0];           // (16,128,8192)
  const float* cw = (const float*)d_in[1];           // (256,128,9,5)
  const float* op = (const float*)d_in[2];           // (256,5)
  float* out = (float*)d_out;                        // (16,256,8192)

  unsigned short* wsA = (unsigned short*)d_ws;                        // 2,949,120 B
  float* opw = (float*)((char*)d_ws + (size_t)1474560 * sizeof(unsigned short));

  const int wElems = 16 * 9 * 4 * 5 * 512;           // 1,474,560
  prep_weights_kernel<<<(wElems + 255) / 256, 256, 0, stream>>>(cw, wsA);
  prep_softmax_kernel<<<1, 256, 0, stream>>>(op, opw);

  dim3 grid(64, 16, 16);   // t-blocks, co-tiles, batches
  selfonn_main_kernel<<<grid, 256, 0, stream>>>(x, wsA, opw, out);
}
